// MulRel_57638461112390
// MI455X (gfx1250) — matrix operand order repaired and checked by reading, not compiled, not run on hardware
//
#include <hip/hip_runtime.h>

// ---------------------------------------------------------------------------
// MulRel fused attention-like op for MI455X (gfx1250, wave32, WMMA).
// All heavy math runs on v_wmma_f32_16x16x32_bf16 (f32 accumulate).
// ---------------------------------------------------------------------------

#define DIM   256
#define RELS  8
#define BATCH 8
#define SEQ   512     // S == T
#define SEQ1  513     // with appended none-node row
#define SP    544     // padded to a multiple of 32
#define TP    544
#define TBLK  16      // t-columns handled per softmax workgroup

typedef __bf16 bf16_t;
typedef __attribute__((ext_vector_type(16))) __bf16 v16bf;
typedef __attribute__((ext_vector_type(8)))  __bf16 v8bf;
typedef __attribute__((ext_vector_type(8)))  float  v8f;

__device__ __forceinline__ unsigned short f32_to_bf16_bits(float f) {
  unsigned int x = __builtin_bit_cast(unsigned int, f);
  x += 0x7fffu + ((x >> 16) & 1u);          // round-to-nearest-even
  return (unsigned short)(x >> 16);
}

// A fragment: 16x32 (MxK) bf16 from row-major [M x K], leading dim ld.
// ISA layout: lanes 0-15 -> K{0..7,16..23}, lanes 16-31 -> K{8..15,24..31}.
__device__ __forceinline__ v16bf load_a_frag(const bf16_t* p, int ld) {
  const int lane = threadIdx.x & 31;
  const int m    = lane & 15;
  const int k0   = (lane >> 4) << 3;        // 0 or 8
  const bf16_t* q = p + (size_t)m * ld + k0;
  v8bf lo = *(const v8bf*)(q);              // K = k0 .. k0+7
  v8bf hi = *(const v8bf*)(q + 16);         // K = k0+16 .. k0+23
  return __builtin_shufflevector(lo, hi, 0,1,2,3,4,5,6,7,8,9,10,11,12,13,14,15);
}

// B fragment: 32x16 (KxN) bf16; source is row-major [N x K] (rows are B columns).
// ISA layout: lanes 0-15 -> K{0..7,16..23} of column n, lanes 16-31 -> K{8..15,24..31}.
__device__ __forceinline__ v16bf load_b_frag(const bf16_t* p, int ld) {
  const int lane = threadIdx.x & 31;
  const int n    = lane & 15;
  const int k0   = (lane >> 4) << 3;        // 0 or 8
  const bf16_t* q = p + (size_t)n * ld + k0;
  v8bf lo = *(const v8bf*)(q);
  v8bf hi = *(const v8bf*)(q + 16);
  return __builtin_shufflevector(lo, hi, 0,1,2,3,4,5,6,7,8,9,10,11,12,13,14,15);
}

__device__ __forceinline__ v8f wmma_bf16(v16bf a, v16bf b, v8f c) {
  return __builtin_amdgcn_wmma_f32_16x16x32_bf16(false, a, false, b, (short)0, c,
                                                 false, false);
}

// C/D layout: col = lane&15, row = vgpr + 8*(lane>>4).
__device__ __forceinline__ void store_tile_bf16(unsigned short* C, int ldc, v8f acc) {
  const int lane = threadIdx.x & 31;
  const int col  = lane & 15;
  const int r0   = (lane >> 4) << 3;
#pragma unroll
  for (int j = 0; j < 8; ++j)
    C[(size_t)(r0 + j) * ldc + col] = f32_to_bf16_bits(acc[j]);
}

__device__ __forceinline__ void store_tile_f32(float* C, int ldc, v8f acc, float scale) {
  const int lane = threadIdx.x & 31;
  const int col  = lane & 15;
  const int r0   = (lane >> 4) << 3;
#pragma unroll
  for (int j = 0; j < 8; ++j)
    C[(size_t)(r0 + j) * ldc + col] = acc[j] * scale;
}

// ---------------------------------------------------------------------------
// Stage 0a: pack embeddings -> bf16, append none-node row at s==512, zero-pad.
// ---------------------------------------------------------------------------
__global__ void pack_emb_kernel(const float* __restrict__ emb,
                                const float* __restrict__ none_node,
                                unsigned short* __restrict__ out, int total) {
  for (int i = blockIdx.x * blockDim.x + threadIdx.x; i < total;
       i += gridDim.x * blockDim.x) {
    int d    = i & (DIM - 1);
    int rest = i >> 8;                       // / DIM
    int s    = rest % SP;
    int b    = rest / SP;
    float v;
    if (s < SEQ)        v = emb[((size_t)b * SEQ + s) * DIM + d];
    else if (s == SEQ)  v = none_node[d];
    else                v = 0.0f;
    out[i] = f32_to_bf16_bits(v);
  }
}

// Stage 0b: transpose weights (r,d,e)->(r,e,d) in bf16 so B-operand loads are
// row-major.
__global__ void transpose_w_kernel(const float* __restrict__ w,
                                   unsigned short* __restrict__ wT, int total) {
  for (int i = blockIdx.x * blockDim.x + threadIdx.x; i < total;
       i += gridDim.x * blockDim.x) {
    int d    = i & 255;
    int rest = i >> 8;
    int e    = rest & 255;
    int r    = rest >> 8;
    wT[i] = f32_to_bf16_bits(w[((size_t)r * DIM + d) * DIM + e]);
  }
}

// ---------------------------------------------------------------------------
// Stage 1: scores[r,b] (SP x DIM) = srcb[b] (SP x DIM) @ Wscore[r] (DIM x DIM)
// B-operand from wscoreT (DIM_out rows x DIM_in), row-major.
// grid = (64 batches, SP/32); 8 waves cover N = 256.
// ---------------------------------------------------------------------------
__global__ void gemm_scores_kernel(const unsigned short* __restrict__ srcb_,
                                   const unsigned short* __restrict__ wT_,
                                   unsigned short* __restrict__ scores) {
  const bf16_t* srcb = (const bf16_t*)srcb_;
  const bf16_t* wT   = (const bf16_t*)wT_;
  const int rb = blockIdx.x, r = rb >> 3, b = rb & 7;
  const int m0 = blockIdx.y * 32;
  const int n0 = (threadIdx.x >> 5) * 32;
  const bf16_t* A  = srcb + (size_t)b * SP * DIM + (size_t)m0 * DIM;
  const bf16_t* Bm = wT + (size_t)r * DIM * DIM + (size_t)n0 * DIM;
  v8f acc[2][2] = {};
  for (int k = 0; k < DIM; k += 32) {
    v16bf a0 = load_a_frag(A + k, DIM);
    v16bf a1 = load_a_frag(A + (size_t)16 * DIM + k, DIM);
    v16bf b0 = load_b_frag(Bm + k, DIM);
    v16bf b1 = load_b_frag(Bm + (size_t)16 * DIM + k, DIM);
    acc[0][0] = wmma_bf16(a0, b0, acc[0][0]);
    acc[0][1] = wmma_bf16(a0, b1, acc[0][1]);
    acc[1][0] = wmma_bf16(a1, b0, acc[1][0]);
    acc[1][1] = wmma_bf16(a1, b1, acc[1][1]);
  }
  unsigned short* C = scores + (size_t)rb * SP * DIM + (size_t)m0 * DIM + n0;
#pragma unroll
  for (int i = 0; i < 2; ++i)
#pragma unroll
    for (int j = 0; j < 2; ++j)
      store_tile_bf16(C + (size_t)i * 16 * DIM + j * 16, DIM, acc[i][j]);
}

// ---------------------------------------------------------------------------
// Stage 2: reltarT[r,b] (DIM x TP) = WembT[r] (DIM x DIM) @ tarb[b]^T
//   reltarT[dout,t] = sum_din WembT[dout,din] * tarb[t,din]  == rel_tar[t,dout]
// grid = (64 batches, TP/32); 8 waves cover M = 256.
// ---------------------------------------------------------------------------
__global__ void gemm_reltar_kernel(const unsigned short* __restrict__ wT_,
                                   const unsigned short* __restrict__ tarb_,
                                   unsigned short* __restrict__ reltarT) {
  const bf16_t* wT   = (const bf16_t*)wT_;
  const bf16_t* tarb = (const bf16_t*)tarb_;
  const int rb = blockIdx.x, r = rb >> 3, b = rb & 7;
  const int n0 = blockIdx.y * 32;                 // t
  const int m0 = (threadIdx.x >> 5) * 32;         // dout
  const bf16_t* A  = wT + (size_t)r * DIM * DIM + (size_t)m0 * DIM;
  const bf16_t* Bm = tarb + (size_t)b * TP * DIM + (size_t)n0 * DIM;
  v8f acc[2][2] = {};
  for (int k = 0; k < DIM; k += 32) {
    v16bf a0 = load_a_frag(A + k, DIM);
    v16bf a1 = load_a_frag(A + (size_t)16 * DIM + k, DIM);
    v16bf b0 = load_b_frag(Bm + k, DIM);
    v16bf b1 = load_b_frag(Bm + (size_t)16 * DIM + k, DIM);
    acc[0][0] = wmma_bf16(a0, b0, acc[0][0]);
    acc[0][1] = wmma_bf16(a0, b1, acc[0][1]);
    acc[1][0] = wmma_bf16(a1, b0, acc[1][0]);
    acc[1][1] = wmma_bf16(a1, b1, acc[1][1]);
  }
  unsigned short* C = reltarT + (size_t)rb * DIM * TP + (size_t)m0 * TP + n0;
#pragma unroll
  for (int i = 0; i < 2; ++i)
#pragma unroll
    for (int j = 0; j < 2; ++j)
      store_tile_bf16(C + (size_t)i * 16 * TP + j * 16, TP, acc[i][j]);
}

// ---------------------------------------------------------------------------
// Stage 3: fused st-GEMM + mask/scale + column softmax (over s).
// One WG per (r, b, 16-column t-block); full 544x16 logit block in LDS.
// st[s,t] = sum_e scores[s,e] * tarb[t,e]; w = softmax_s((st+mask)/16).
// ---------------------------------------------------------------------------
__global__ void st_softmax_kernel(const unsigned short* __restrict__ scores_,
                                  const unsigned short* __restrict__ tarb_,
                                  const float* __restrict__ src_mask,
                                  const float* __restrict__ tar_mask,
                                  unsigned short* __restrict__ wsoft) {
  __shared__ float stb[SP * TBLK];          // 34,816 B
  __shared__ float red[256];
  __shared__ float colmax[TBLK];
  __shared__ float colrcp[TBLK];

  const bf16_t* scores = (const bf16_t*)scores_;
  const bf16_t* tarb   = (const bf16_t*)tarb_;
  const int tb = blockIdx.x;                // 0..31 -> t0 < 512 only
  const int b  = blockIdx.y;
  const int r  = blockIdx.z;
  const int rb = r * BATCH + b;
  const int t0 = tb * TBLK;
  const int wave = threadIdx.x >> 5;
  const int lane = threadIdx.x & 31;
  const int col  = lane & 15;
  const int r0   = (lane >> 4) << 3;

  // Hoist all 8 K-step B fragments (16 target rows) into registers.
  const bf16_t* Brow = tarb + (size_t)b * TP * DIM + (size_t)t0 * DIM;
  v16bf bfr[8];
#pragma unroll
  for (int k = 0; k < 8; ++k) bfr[k] = load_b_frag(Brow + k * 32, DIM);

  const bf16_t* A  = scores + (size_t)rb * SP * DIM;
  const float   tm = tar_mask[(size_t)b * SEQ + t0 + col];

  for (int mt = wave; mt < SP / 16; mt += 8) {
    v8f acc = {};
    const bf16_t* Am = A + (size_t)mt * 16 * DIM;
#pragma unroll
    for (int k = 0; k < 8; ++k) {
      v16bf a = load_a_frag(Am + k * 32, DIM);
      acc = wmma_bf16(a, bfr[k], acc);
    }
#pragma unroll
    for (int j = 0; j < 8; ++j) {
      const int srow = mt * 16 + r0 + j;
      float v;
      if (srow >= SEQ1) {
        v = -1e30f;                         // padded rows excluded from softmax
      } else {
        const float sm = (srow < SEQ) ? src_mask[(size_t)b * SEQ + srow] : 1.0f;
        v = (acc[j] + (1.0f - sm * tm) * -1e10f) * 0.0625f;   // 1/sqrt(256)
      }
      stb[srow * TBLK + col] = v;
    }
  }
  __syncthreads();

  // Column-wise softmax over 544 rows; 16 partials per column.
  const int scol = threadIdx.x & (TBLK - 1);
  const int part = threadIdx.x >> 4;        // 0..15
  const int rows_per = SP / 16;             // 34
  float m = -3.4e38f;
  for (int i = 0; i < rows_per; ++i)
    m = fmaxf(m, stb[(part * rows_per + i) * TBLK + scol]);
  red[threadIdx.x] = m;
  __syncthreads();
  if (threadIdx.x < TBLK) {
    float mm = -3.4e38f;
    for (int p = 0; p < 16; ++p) mm = fmaxf(mm, red[p * TBLK + threadIdx.x]);
    colmax[threadIdx.x] = mm;
  }
  __syncthreads();
  const float cm = colmax[scol];
  float s = 0.0f;
  for (int i = 0; i < rows_per; ++i) {
    const int idx = (part * rows_per + i) * TBLK + scol;
    const float e = __expf(stb[idx] - cm);
    stb[idx] = e;
    s += e;
  }
  red[threadIdx.x] = s;
  __syncthreads();
  if (threadIdx.x < TBLK) {
    float ss = 0.0f;
    for (int p = 0; p < 16; ++p) ss += red[p * TBLK + threadIdx.x];
    colrcp[threadIdx.x] = 1.0f / ss;
  }
  __syncthreads();
  for (int f = threadIdx.x; f < SP * TBLK; f += 256) {
    const int row = f >> 4;
    const int c   = f & 15;
    const float w = stb[f] * colrcp[c];
    wsoft[((size_t)rb * SP + row) * TP + t0 + c] = f32_to_bf16_bits(w);
  }
}

// ---------------------------------------------------------------------------
// Stage 4: out[b,s,d] = (1/sqrt(8)) * sum_r sum_{t<512} w[r,b,s,t]*reltarT[r,b,d,t]
// grid = (B, 512/32); 8 waves cover D = 256. f32 output.
// ---------------------------------------------------------------------------
__global__ void out_gemm_kernel(const unsigned short* __restrict__ wsoft_,
                                const unsigned short* __restrict__ reltarT_,
                                float* __restrict__ out) {
  const bf16_t* wsoft = (const bf16_t*)wsoft_;
  const bf16_t* rt    = (const bf16_t*)reltarT_;
  const int b  = blockIdx.x;
  const int m0 = blockIdx.y * 32;                 // s
  const int n0 = (threadIdx.x >> 5) * 32;         // d
  v8f acc[2][2] = {};
  for (int r = 0; r < RELS; ++r) {
    const int rb = r * BATCH + b;
    const bf16_t* A  = wsoft + (size_t)rb * SP * TP + (size_t)m0 * TP;
    const bf16_t* Bm = rt + (size_t)rb * DIM * TP + (size_t)n0 * TP;
    for (int k = 0; k < SEQ; k += 32) {           // only t < 512 contributes
      v16bf a0 = load_a_frag(A + k, TP);
      v16bf a1 = load_a_frag(A + (size_t)16 * TP + k, TP);
      v16bf b0 = load_b_frag(Bm + k, TP);
      v16bf b1 = load_b_frag(Bm + (size_t)16 * TP + k, TP);
      acc[0][0] = wmma_bf16(a0, b0, acc[0][0]);
      acc[0][1] = wmma_bf16(a0, b1, acc[0][1]);
      acc[1][0] = wmma_bf16(a1, b0, acc[1][0]);
      acc[1][1] = wmma_bf16(a1, b1, acc[1][1]);
    }
  }
  const float scale = 0.35355339059327373f;       // 1/sqrt(REL)
  float* C = out + ((size_t)b * SEQ + m0) * DIM + n0;
#pragma unroll
  for (int i = 0; i < 2; ++i)
#pragma unroll
    for (int j = 0; j < 2; ++j)
      store_tile_f32(C + (size_t)i * 16 * DIM + j * 16, DIM, acc[i][j], scale);
}

// ---------------------------------------------------------------------------
extern "C" void kernel_launch(void* const* d_in, const int* in_sizes, int n_in,
                              void* d_out, int out_size, void* d_ws, size_t ws_size,
                              hipStream_t stream) {
  const float* src_emb  = (const float*)d_in[0];
  const float* tar_emb  = (const float*)d_in[1];
  const float* src_mask = (const float*)d_in[2];
  const float* tar_mask = (const float*)d_in[3];
  const float* w_score  = (const float*)d_in[4];
  const float* w_emb    = (const float*)d_in[5];
  const float* src_none = (const float*)d_in[6];
  const float* tar_none = (const float*)d_in[7];

  char* ws = (char*)d_ws;
  size_t o = 0;
  unsigned short* srcb    = (unsigned short*)(ws + o); o += (size_t)BATCH * SP * DIM * 2;
  unsigned short* tarb    = (unsigned short*)(ws + o); o += (size_t)BATCH * TP * DIM * 2;
  unsigned short* wscoreT = (unsigned short*)(ws + o); o += (size_t)RELS * DIM * DIM * 2;
  unsigned short* wembT   = (unsigned short*)(ws + o); o += (size_t)RELS * DIM * DIM * 2;
  unsigned short* scores  = (unsigned short*)(ws + o); o += (size_t)RELS * BATCH * SP * DIM * 2;
  unsigned short* reltarT = (unsigned short*)(ws + o); o += (size_t)RELS * BATCH * DIM * TP * 2;
  unsigned short* wsoft   = (unsigned short*)(ws + o); o += (size_t)RELS * BATCH * SP * TP * 2;
  (void)ws_size; (void)in_sizes; (void)n_in; (void)out_size;

  const dim3 blk(256);
  pack_emb_kernel<<<1024, blk, 0, stream>>>(src_emb, src_none, srcb, BATCH * SP * DIM);
  pack_emb_kernel<<<1024, blk, 0, stream>>>(tar_emb, tar_none, tarb, BATCH * TP * DIM);
  transpose_w_kernel<<<512, blk, 0, stream>>>(w_score, wscoreT, RELS * DIM * DIM);
  transpose_w_kernel<<<512, blk, 0, stream>>>(w_emb, wembT, RELS * DIM * DIM);

  gemm_scores_kernel<<<dim3(RELS * BATCH, SP / 32), blk, 0, stream>>>(srcb, wscoreT, scores);
  gemm_reltar_kernel<<<dim3(RELS * BATCH, TP / 32), blk, 0, stream>>>(wembT, tarb, reltarT);
  st_softmax_kernel<<<dim3(SEQ / TBLK, BATCH, RELS), blk, 0, stream>>>(
      scores, tarb, src_mask, tar_mask, wsoft);
  out_gemm_kernel<<<dim3(BATCH, SEQ / 32), blk, 0, stream>>>(wsoft, reltarT, (float*)d_out);
}
